// AttentionBlock_52982716563530
// MI455X (gfx1250) — matrix operand order repaired and checked by reading, not compiled, not run on hardware
//
#include <hip/hip_runtime.h>

// ---------------------------------------------------------------------------
// AttentionBlock for MI455X (gfx1250, wave32, WMMA).
// All GEMMs use v_wmma_f32_16x16x32_f16 (f16 in, f32 accumulate).
// Attention is fused (flash-style): no 128MB score matrix round-trip.
// The c^-0.5 score scale is folded into q at the QKV epilogue.
// ---------------------------------------------------------------------------

typedef __attribute__((ext_vector_type(16))) _Float16 v16h;
typedef __attribute__((ext_vector_type(8)))  _Float16 v8h;
typedef __attribute__((ext_vector_type(4)))  _Float16 v4h;
typedef __attribute__((ext_vector_type(8)))  float    v8f;

#define NB   32          // batch
#define CC   512         // channels
#define HW   1024        // h*w
#define OC   1536        // 3*C

// ---- WMMA wrapper ---------------------------------------------------------
__device__ __forceinline__ v8f wmma16(v16h a, v16h b, v8f c) {
  // 8 args: (neg_a, A, neg_b, B, c_mod, C, reuse_a, reuse_b)
  return __builtin_amdgcn_wmma_f32_16x16x32_f16(false, a, false, b,
                                                (short)0, c, false, false);
}

__device__ __forceinline__ v8f zero8() {
  v8f z;
#pragma unroll
  for (int i = 0; i < 8; ++i) z[i] = 0.f;
  return z;
}

union AFrag { v16h v; v8h h[2]; };

// A-matrix 16x32 f16 fragment (row-major A, leading dim lda, in elements).
// Per ISA 7.12.2: lane<16 -> M=lane, K in {0..7},{16..23};
//                 lane>=16 -> M=lane-16, K in {8..15},{24..31}.
__device__ __forceinline__ v16h load_a(const _Float16* A, int lda, int lane) {
  const int m  = lane & 15;
  const int ks = (lane >> 4) << 3;          // 0 or 8
  const _Float16* p = A + (size_t)m * lda + ks;
  AFrag u;
  u.h[0] = *(const v8h*)(p);
  u.h[1] = *(const v8h*)(p + 16);
  return u.v;
}

// B-matrix 32x16 f16 fragment (K-major B, leading dim ldb).
// Per lane: column N=lane%16, K in {8h..8h+7},{16+8h..16+8h+7}, h=lane/16.
__device__ __forceinline__ v16h load_b(const _Float16* Bp, int ldb, int lane) {
  const int n  = lane & 15;
  const int ks = (lane >> 4) << 3;
  v16h r;
#pragma unroll
  for (int s = 0; s < 8; ++s) {
    r[s]     = Bp[(size_t)(ks + s) * ldb + n];
    r[s + 8] = Bp[(size_t)(16 + ks + s) * ldb + n];
  }
  return r;
}

// ---- Kernel 1: f32 -> f16 conversion --------------------------------------
__global__ void cvt_f16_kernel(const float* __restrict__ in,
                               _Float16* __restrict__ out, int n) {
  int i = blockIdx.x * blockDim.x + threadIdx.x;
  if (i < n) out[i] = (_Float16)in[i];
}

// ---- Kernel 2: GroupNorm, write xn as f16 [b][c][hw] ----------------------
__global__ __launch_bounds__(256) void gn_kernel(const float* __restrict__ x,
                                                 const float* __restrict__ gw,
                                                 const float* __restrict__ gb,
                                                 _Float16* __restrict__ xn) {
  const int g = blockIdx.x, b = blockIdx.y;           // grid (32, 32)
  const int tid = threadIdx.x;
  const size_t base = ((size_t)b * CC + g * 16) * HW; // 16 ch * 1024 = 16384 f32
  const float4* xp = (const float4*)(x + base);       // 4096 float4

  float4 loc[16];
  float s = 0.f, s2 = 0.f;
#pragma unroll
  for (int i = 0; i < 16; ++i) {
    float4 t = xp[tid + i * 256];
    loc[i] = t;
    s  += t.x + t.y + t.z + t.w;
    s2 += t.x * t.x + t.y * t.y + t.z * t.z + t.w * t.w;
  }
  __shared__ float rs[256], rq[256];
  rs[tid] = s; rq[tid] = s2;
  __syncthreads();
  for (int st = 128; st > 0; st >>= 1) {
    if (tid < st) { rs[tid] += rs[tid + st]; rq[tid] += rq[tid + st]; }
    __syncthreads();
  }
  const float mean = rs[0] * (1.f / 16384.f);
  const float var  = rq[0] * (1.f / 16384.f) - mean * mean;
  const float rstd = rsqrtf(var + 1e-5f);

  v4h* op = (v4h*)(xn + base);
#pragma unroll
  for (int i = 0; i < 16; ++i) {
    int idx4 = tid + i * 256;
    int c = g * 16 + (idx4 >> 8);             // 256 float4 per channel
    float ga = gw[c] * rstd;
    float be = gb[c] - mean * ga;
    float4 t = loc[i];
    v4h o;
    o[0] = (_Float16)(t.x * ga + be);
    o[1] = (_Float16)(t.y * ga + be);
    o[2] = (_Float16)(t.z * ga + be);
    o[3] = (_Float16)(t.w * ga + be);
    op[idx4] = o;
  }
}

// ---- Kernel 3: QKV GEMM per batch -----------------------------------------
// C[o,n] = sum_c W[o,c] * xn[b,c,n] + bias[o],  o in [0,1536), n in [0,1024)
// Scatter: q[b][n][c] (pre-scaled by 512^-0.5), k[b][c][n], v[b][n][c]  (f16)
// grid (12, 4, 32) : 128(o) x 256(n) macro-tile, 8 waves of 64x64.
__global__ __launch_bounds__(256) void qkv_kernel(
    const _Float16* __restrict__ W, const _Float16* __restrict__ xn,
    const float* __restrict__ bias, _Float16* __restrict__ q,
    _Float16* __restrict__ k, _Float16* __restrict__ v) {
  const int lane = threadIdx.x & 31;
  const int wv   = threadIdx.x >> 5;          // 0..7
  const int b    = blockIdx.z;
  const int o0   = blockIdx.x * 128 + (wv >> 2) * 64;
  const int n0   = blockIdx.y * 256 + (wv & 3) * 64;
  const float qscale = 0.044194173824159216f; // 512^-0.5 folded into q

  const _Float16* A  = W + (size_t)o0 * CC;
  const _Float16* Bp = xn + (size_t)b * CC * HW + n0;

  v8f acc[4][4];
#pragma unroll
  for (int i = 0; i < 4; ++i)
#pragma unroll
    for (int j = 0; j < 4; ++j) acc[i][j] = zero8();

  for (int kk = 0; kk < CC; kk += 32) {
    v16h af[4], bf[4];
#pragma unroll
    for (int i = 0; i < 4; ++i) af[i] = load_a(A + (size_t)i * 16 * CC + kk, CC, lane);
#pragma unroll
    for (int j = 0; j < 4; ++j) bf[j] = load_b(Bp + (size_t)kk * HW + j * 16, HW, lane);
#pragma unroll
    for (int i = 0; i < 4; ++i)
#pragma unroll
      for (int j = 0; j < 4; ++j) acc[i][j] = wmma16(af[i], bf[j], acc[i][j]);
  }

  const int region = o0 >> 9;                  // 0:q 1:k 2:v (uniform per wave)
  const int hi = (lane >> 4) << 3;             // 0 or 8
#pragma unroll
  for (int mt = 0; mt < 4; ++mt) {
    const int ob = o0 + mt * 16 + hi;          // first of 8 consecutive o rows
#pragma unroll
    for (int nt = 0; nt < 4; ++nt) {
      const int n = n0 + nt * 16 + (lane & 15);
      if (region == 0) {
        v8h pk;
#pragma unroll
        for (int r = 0; r < 8; ++r)
          pk[r] = (_Float16)((acc[mt][nt][r] + bias[ob + r]) * qscale);
        *(v8h*)&q[((size_t)b * HW + n) * CC + ob] = pk;             // [n][c]
      } else if (region == 2) {
        v8h pk;
#pragma unroll
        for (int r = 0; r < 8; ++r)
          pk[r] = (_Float16)(acc[mt][nt][r] + bias[ob + r]);
        *(v8h*)&v[((size_t)b * HW + n) * CC + (ob - 1024)] = pk;    // [m][c]
      } else {
#pragma unroll
        for (int r = 0; r < 8; ++r)                                  // [c][m]
          k[((size_t)b * CC + (ob - 512 + r)) * HW + n] =
              (_Float16)(acc[mt][nt][r] + bias[ob + r]);
      }
    }
  }
}

// ---- Kernel 4: fused flash attention --------------------------------------
// Each wave: 16 query rows, full d=512. O accum = 32 tiles (256 VGPRs, f32).
// P (16x64) goes C-layout -> A-layout through per-wave LDS.
// ao[b][c][n] f16 (c-major: exactly the B layout the projection GEMM wants).
// grid (16, 32), block 128 (4 independent waves).
__global__ __launch_bounds__(128, 1) void flash_kernel(
    const _Float16* __restrict__ q, const _Float16* __restrict__ k,
    const _Float16* __restrict__ v, _Float16* __restrict__ ao) {
  __shared__ __align__(32) _Float16 Plds[4][16 * 64];
  const int lane = threadIdx.x & 31;
  const int wv   = threadIdx.x >> 5;
  const int b    = blockIdx.y;
  const int n0   = blockIdx.x * 64 + wv * 16;

  const _Float16* qb = q + (size_t)b * HW * CC;   // [n][c] (pre-scaled)
  const _Float16* kb = k + (size_t)b * CC * HW;   // [c][m]
  const _Float16* vb = v + (size_t)b * HW * CC;   // [m][c]

  v8f O[32];
#pragma unroll
  for (int i = 0; i < 32; ++i) O[i] = zero8();
  float rmax[8], rsum[8];
#pragma unroll
  for (int r = 0; r < 8; ++r) { rmax[r] = -3.0e38f; rsum[r] = 0.f; }

  const int hi   = (lane >> 4) << 3;           // row-half offset for C tiles
  const int mrow = lane & 15;                  // A-fragment row for P

  for (int m0 = 0; m0 < HW; m0 += 64) {
    // --- S = Q[n0:16,:] x K[:, m0:64]  (q pre-scaled by 512^-0.5) ---
    v8f S[4];
#pragma unroll
    for (int t = 0; t < 4; ++t) S[t] = zero8();
    for (int kk = 0; kk < CC; kk += 32) {
      v16h af = load_a(qb + (size_t)n0 * CC + kk, CC, lane);
#pragma unroll
      for (int t = 0; t < 4; ++t) {
        v16h bf = load_b(kb + (size_t)kk * HW + m0 + t * 16, HW, lane);
        S[t] = wmma16(af, bf, S[t]);
      }
    }
    // --- online softmax (row stats per lane-half via shfl_xor <16) ---
    float tm[8];
#pragma unroll
    for (int r = 0; r < 8; ++r) {
      float m01 = fmaxf(S[0][r], S[1][r]);
      float m23 = fmaxf(S[2][r], S[3][r]);
      tm[r] = fmaxf(m01, m23);
    }
#pragma unroll
    for (int mask = 1; mask < 16; mask <<= 1)
#pragma unroll
      for (int r = 0; r < 8; ++r) tm[r] = fmaxf(tm[r], __shfl_xor(tm[r], mask, 32));

    float sc[8];
#pragma unroll
    for (int r = 0; r < 8; ++r) {
      float nm = fmaxf(rmax[r], tm[r]);
      sc[r] = __expf(rmax[r] - nm);
      rmax[r] = nm;
    }
    float ps[8];
#pragma unroll
    for (int r = 0; r < 8; ++r) ps[r] = 0.f;
#pragma unroll
    for (int t = 0; t < 4; ++t) {
      const int col = t * 16 + (lane & 15);
#pragma unroll
      for (int r = 0; r < 8; ++r) {
        float p = __expf(S[t][r] - rmax[r]);
        ps[r] += p;
        Plds[wv][(r + hi) * 64 + col] = (_Float16)p;     // ds_store_b16
      }
    }
#pragma unroll
    for (int mask = 1; mask < 16; mask <<= 1)
#pragma unroll
      for (int r = 0; r < 8; ++r) ps[r] += __shfl_xor(ps[r], mask, 32);
#pragma unroll
    for (int r = 0; r < 8; ++r) rsum[r] = rsum[r] * sc[r] + ps[r];

    // rescale O by exp(old_max - new_max)
#pragma unroll
    for (int dt = 0; dt < 32; ++dt)
#pragma unroll
      for (int r = 0; r < 8; ++r) O[dt][r] *= sc[r];

    // cross-lane LDS dependency within this wave: split-counter wait (CDNA5)
    asm volatile("s_wait_dscnt 0" ::: "memory");

    // --- O += P x V[m0:64, :] ---
#pragma unroll
    for (int ks = 0; ks < 2; ++ks) {
      AFrag u;                                           // P as A-fragment (ds_load)
      const v8h* lp = (const v8h*)&Plds[wv][mrow * 64 + ks * 32 + hi];
      u.h[0] = lp[0];
      u.h[1] = lp[2];                                    // +16 halves = +32 bytes
      v16h pa = u.v;
#pragma unroll
      for (int dt = 0; dt < 32; ++dt) {
        v16h bf = load_b(vb + (size_t)(m0 + ks * 32) * CC + dt * 16, CC, lane);
        O[dt] = wmma16(pa, bf, O[dt]);
      }
    }
  }

  // --- normalize and store ao[b][c][n] (contiguous b128 stores) ---
  float inv[8];
#pragma unroll
  for (int r = 0; r < 8; ++r) inv[r] = 1.f / rsum[r];
#pragma unroll
  for (int dt = 0; dt < 32; ++dt) {
    const int c = dt * 16 + (lane & 15);
    v8h o;
#pragma unroll
    for (int r = 0; r < 8; ++r) o[r] = (_Float16)(O[dt][r] * inv[r]);
    *(v8h*)&ao[((size_t)b * CC + c) * HW + n0 + hi] = o;
  }
}

// ---- Kernel 5: output projection + bias + residual ------------------------
// y[b][o][n] = sum_c Wo[o,c]*ao[b,c,n] + ob[o] + x[b][o][n]   (f32 out)
// grid (4, 4, 32)
__global__ __launch_bounds__(256) void proj_kernel(
    const _Float16* __restrict__ Wo, const _Float16* __restrict__ ao,
    const float* __restrict__ obias, const float* __restrict__ x,
    float* __restrict__ y) {
  const int lane = threadIdx.x & 31;
  const int wv   = threadIdx.x >> 5;
  const int b    = blockIdx.z;
  const int o0   = blockIdx.x * 128 + (wv >> 2) * 64;
  const int n0   = blockIdx.y * 256 + (wv & 3) * 64;

  const _Float16* A  = Wo + (size_t)o0 * CC;
  const _Float16* Bp = ao + (size_t)b * CC * HW + n0;

  v8f acc[4][4];
#pragma unroll
  for (int i = 0; i < 4; ++i)
#pragma unroll
    for (int j = 0; j < 4; ++j) acc[i][j] = zero8();

  for (int kk = 0; kk < CC; kk += 32) {
    v16h af[4], bf[4];
#pragma unroll
    for (int i = 0; i < 4; ++i) af[i] = load_a(A + (size_t)i * 16 * CC + kk, CC, lane);
#pragma unroll
    for (int j = 0; j < 4; ++j) bf[j] = load_b(Bp + (size_t)kk * HW + j * 16, HW, lane);
#pragma unroll
    for (int i = 0; i < 4; ++i)
#pragma unroll
      for (int j = 0; j < 4; ++j) acc[i][j] = wmma16(af[i], bf[j], acc[i][j]);
  }

  const int hi = (lane >> 4) << 3;
#pragma unroll
  for (int mt = 0; mt < 4; ++mt) {
    const int ob = o0 + mt * 16 + hi;
#pragma unroll
    for (int nt = 0; nt < 4; ++nt) {
      const int n = n0 + nt * 16 + (lane & 15);
#pragma unroll
      for (int r = 0; r < 8; ++r) {
        const size_t idx = ((size_t)b * CC + (ob + r)) * HW + n;
        y[idx] = acc[mt][nt][r] + obias[ob + r] + x[idx];
      }
    }
  }
}

// ---------------------------------------------------------------------------
extern "C" void kernel_launch(void* const* d_in, const int* in_sizes, int n_in,
                              void* d_out, int out_size, void* d_ws, size_t ws_size,
                              hipStream_t stream) {
  (void)in_sizes; (void)n_in; (void)out_size; (void)ws_size;

  const float* x     = (const float*)d_in[0];  // [32,512,32,32]
  const float* gn_w  = (const float*)d_in[1];  // [512]
  const float* gn_b  = (const float*)d_in[2];  // [512]
  const float* qkv_w = (const float*)d_in[3];  // [1536,512]
  const float* qkv_b = (const float*)d_in[4];  // [1536]
  const float* out_w = (const float*)d_in[5];  // [512,512]
  const float* out_b = (const float*)d_in[6];  // [512]
  float* y = (float*)d_out;

  // workspace carving (≈170 MB: 5 x 33.5MB f16 tensors + f16 weights)
  char* w = (char*)d_ws;
  const size_t T = (size_t)NB * CC * HW * sizeof(_Float16);   // 33,554,432 B
  _Float16* xn   = (_Float16*)(w);           // [b][c][hw]
  _Float16* qbuf = (_Float16*)(w + T);       // [b][n][c]
  _Float16* kbuf = (_Float16*)(w + 2 * T);   // [b][c][m]
  _Float16* vbuf = (_Float16*)(w + 3 * T);   // [b][m][c]
  _Float16* ao   = (_Float16*)(w + 4 * T);   // [b][c][n]
  _Float16* wq16 = (_Float16*)(w + 5 * T);                     // 1.5 MB
  _Float16* wo16 = (_Float16*)(w + 5 * T + (size_t)OC * CC * 2);

  cvt_f16_kernel<<<(OC * CC) / 256, 256, 0, stream>>>(qkv_w, wq16, OC * CC);
  cvt_f16_kernel<<<(CC * CC) / 256, 256, 0, stream>>>(out_w, wo16, CC * CC);
  gn_kernel<<<dim3(32, NB), 256, 0, stream>>>(x, gn_w, gn_b, xn);
  qkv_kernel<<<dim3(12, 4, NB), 256, 0, stream>>>(wq16, xn, qkv_b,
                                                  qbuf, kbuf, vbuf);
  flash_kernel<<<dim3(16, NB), 128, 0, stream>>>(qbuf, kbuf, vbuf, ao);
  proj_kernel<<<dim3(4, 4, NB), 256, 0, stream>>>(wo16, ao, out_b, x, y);
}
